// SimpleLSTMFixedLenNet_28802050687412
// MI455X (gfx1250) — compile-verified, matrix operand order checked by reading, not run on hardware
//
#include <hip/hip_runtime.h>
#include <hip/hip_bf16.h>

// Problem constants (match reference)
#define VOCAB 50000
#define EMB   128
#define HID   128
#define NCLS  8
#define BATCH 256
#define SEQ   512
#define GATES 512        // 4*HID
#define KTOT  256        // EMB + HID (fused A operand)
#define ROWS  16         // batch rows per workgroup
#define NWG   (BATCH/ROWS)

#define WC_BYTES  ((size_t)GATES * KTOT * 2)        // 256 KB
#define EBF_BYTES ((size_t)VOCAB * EMB * 2)         // 12.8 MB

typedef __attribute__((ext_vector_type(16))) __bf16 v16bf;
typedef __attribute__((ext_vector_type(8)))  __bf16 v8bf;
typedef __attribute__((ext_vector_type(8)))  float  v8f;

__device__ __forceinline__ __bf16 f2bf(float f) {
  unsigned u = __builtin_bit_cast(unsigned, f);
  unsigned r = u + 0x7FFFu + ((u >> 16) & 1u);   // round-to-nearest-even
  unsigned short h = (unsigned short)(r >> 16);
  return __builtin_bit_cast(__bf16, h);
}

// Gate nonlinearities on the recurrent critical path: prefer the CDNA5
// hardware transcendental v_tanh_f32; otherwise exp + v_rcp_f32 (no IEEE
// division sequences).
__device__ __forceinline__ float fast_tanh(float x) {
#if __has_builtin(__builtin_amdgcn_tanhf)
  return __builtin_amdgcn_tanhf(x);
#else
  x = fminf(fmaxf(x, -15.0f), 15.0f);
  float e = __expf(2.0f * x);
  return (e - 1.0f) * __builtin_amdgcn_rcpf(e + 1.0f);
#endif
}
__device__ __forceinline__ float sigm(float x) {
#if __has_builtin(__builtin_amdgcn_tanhf)
  return __fmaf_rn(0.5f, __builtin_amdgcn_tanhf(0.5f * x), 0.5f);
#else
  return __builtin_amdgcn_rcpf(1.0f + __expf(-x));
#endif
}

__device__ __forceinline__ void wait_asynccnt0() {
#if __has_builtin(__builtin_amdgcn_s_wait_asynccnt)
  __builtin_amdgcn_s_wait_asynccnt(0);
#else
  asm volatile("s_wait_asynccnt 0" ::: "memory");
#endif
}

// Pack [W_ih | W_hh] row-major into bf16 Wc[512][256] in workspace.
__global__ void prep_weights(const float* __restrict__ W_ih,
                             const float* __restrict__ W_hh,
                             __bf16* __restrict__ Wc) {
  int gate = blockIdx.x;    // 0..511
  int k    = threadIdx.x;   // 0..255
  float v = (k < EMB) ? W_ih[gate * EMB + k] : W_hh[gate * HID + (k - EMB)];
  Wc[gate * KTOT + k] = f2bf(v);
}

// Convert embedding table to bf16 once (L2-resident; halves gather bytes,
// removes per-step conversion VALU from the recurrent critical path).
__global__ void prep_embed(const float* __restrict__ E, __bf16* __restrict__ Ebf) {
  size_t i = (size_t)blockIdx.x * blockDim.x + threadIdx.x;
  if (i < (size_t)VOCAB * EMB) Ebf[i] = f2bf(E[i]);
}

// One workgroup = 16 batch rows for the whole sequence.
// 8 waves; wave w owns h-columns [16w,16w+16) across all 4 gate blocks.
// USE_EBF: bf16 embedding table in workspace + async global->LDS prefetch.
template <bool USE_EBF>
__global__ __launch_bounds__(256, 1)
void lstm_kernel(const int*    __restrict__ x,
                 const float*  __restrict__ E,
                 const __bf16* __restrict__ Ebf,
                 const __bf16* __restrict__ Wc,
                 const float*  __restrict__ b_ih,
                 const float*  __restrict__ b_hh,
                 const float*  __restrict__ W_l,
                 const float*  __restrict__ b_l,
                 float* __restrict__ out) {
  __shared__ __align__(16) __bf16 EmbBuf[2][ROWS][EMB]; // 8 KB double-buffered emb tile
  __shared__ __align__(16) __bf16 Hbuf[ROWS][HID];      // 4 KB h_t (bf16)
  __shared__ __align__(16) float  Hlds[ROWS][HID];      // 8 KB final h (f32)

  const int tid     = threadIdx.x;
  const int lane    = tid & 31;
  const int w       = tid >> 5;        // wave 0..7
  const int g       = blockIdx.x;      // 0..15
  const int halfsel = lane >> 4;       // which K-half of the 16-bit operand fragment
  const int ln      = lane & 15;       // N column (B) / M row (A) within tile

  // ---- B operand fragments, register resident: 4 gate tiles x 8 K-chunks ----
  // 16-bit operand lane layout (ISA 7.12.2):
  //   lanes 0-15 hold K {0..7,16..23}, lanes 16-31 hold K {8..15,24..31}.
  v16bf Bf[4][8];
  #pragma unroll
  for (int gb = 0; gb < 4; ++gb) {
    const __bf16* wrow = Wc + (size_t)(gb * 128 + w * 16 + ln) * KTOT;
    #pragma unroll
    for (int c = 0; c < 8; ++c) {
      const int kb = c * 32 + halfsel * 8;
      v8bf lo = *(const v8bf*)(wrow + kb);
      v8bf hi = *(const v8bf*)(wrow + kb + 16);
      v16bf b;
      #pragma unroll
      for (int i = 0; i < 8; ++i) { b[i] = lo[i]; b[i + 8] = hi[i]; }
      Bf[gb][c] = b;
    }
  }

  // Bias as persistent C fragments (consumed by the peeled first WMMA of each
  // step's K-chain; no per-step accumulator init moves).
  v8f biasf0, biasf1, biasf2, biasf3;
  {
    int n0 = 0 * 128 + w * 16 + ln, n1 = 1 * 128 + w * 16 + ln;
    int n2 = 2 * 128 + w * 16 + ln, n3 = 3 * 128 + w * 16 + ln;
    float bv0 = b_ih[n0] + b_hh[n0], bv1 = b_ih[n1] + b_hh[n1];
    float bv2 = b_ih[n2] + b_hh[n2], bv3 = b_ih[n3] + b_hh[n3];
    #pragma unroll
    for (int e = 0; e < 8; ++e) {
      biasf0[e] = bv0; biasf1[e] = bv1; biasf2[e] = bv2; biasf3[e] = bv3;
    }
  }

  // Staging role: thread -> (row, 8-element segment).
  const int r_stage = tid >> 4;   // 0..15
  const int seg     = tid & 15;   // 0..15
  const long long xrow = (long long)(g * ROWS + r_stage) * SEQ;

  // Per-thread LDS byte offsets of the two emb staging slots (for async copy).
  const unsigned lds_emb0 = (unsigned)(size_t)&EmbBuf[0][r_stage][seg * 8];
  const unsigned lds_emb1 = (unsigned)(size_t)&EmbBuf[1][r_stage][seg * 8];

  // h_0 = 0, c_0 = 0; stage emb_0 into buffer 0 synchronously.
  {
    #pragma unroll
    for (int i = 0; i < 8; ++i) Hbuf[r_stage][seg * 8 + i] = f2bf(0.0f);
    int idx0 = x[xrow];
    if constexpr (USE_EBF) {
      v8bf e0 = *(const v8bf*)(Ebf + (size_t)idx0 * EMB + seg * 8);
      *(v8bf*)&EmbBuf[0][r_stage][seg * 8] = e0;
    } else {
      const float* erow = E + (size_t)idx0 * EMB + seg * 8;
      float4 f0 = *(const float4*)(erow);
      float4 f1 = *(const float4*)(erow + 4);
      v8bf eb;
      eb[0] = f2bf(f0.x); eb[1] = f2bf(f0.y); eb[2] = f2bf(f0.z); eb[3] = f2bf(f0.w);
      eb[4] = f2bf(f1.x); eb[5] = f2bf(f1.y); eb[6] = f2bf(f1.z); eb[7] = f2bf(f1.w);
      *(v8bf*)&EmbBuf[0][r_stage][seg * 8] = eb;
    }
  }

  v8f cst, hf;
  #pragma unroll
  for (int e = 0; e < 8; ++e) { cst[e] = 0.0f; hf[e] = 0.0f; }

  for (int t = 0; t < SEQ; ++t) {
    __syncthreads();  // BARRIER A: emb_t (buf t&1) and h_t visible to all waves

    // ---- prefetch emb_{t+1} into the other buffer while we compute step t ----
    if (t + 1 < SEQ) {
      int nidx = x[xrow + t + 1];
      if constexpr (USE_EBF) {
        unsigned long long gaddr =
            (unsigned long long)(Ebf + (size_t)nidx * EMB + seg * 8);
        unsigned ldst = ((t + 1) & 1) ? lds_emb1 : lds_emb0;
        // CDNA5 async DMA: global -> LDS, tracked by ASYNCcnt (no VGPR data).
        asm volatile("global_load_async_to_lds_b128 %0, %1, off"
                     :: "v"(ldst), "v"(gaddr) : "memory");
      } else {
        const float* erow = E + (size_t)nidx * EMB + seg * 8;
        float4 f0 = *(const float4*)(erow);
        float4 f1 = *(const float4*)(erow + 4);
        v8bf eb;
        eb[0] = f2bf(f0.x); eb[1] = f2bf(f0.y); eb[2] = f2bf(f0.z); eb[3] = f2bf(f0.w);
        eb[4] = f2bf(f1.x); eb[5] = f2bf(f1.y); eb[6] = f2bf(f1.z); eb[7] = f2bf(f1.w);
        *(v8bf*)&EmbBuf[(t + 1) & 1][r_stage][seg * 8] = eb;
      }
    }

    // ---- gates = [emb_t | h_t] @ Wc^T + bias ;  M=16, N=4x16, K=256 ----
    const __bf16* arow_e = &EmbBuf[t & 1][ln][0];
    const __bf16* arow_h = &Hbuf[ln][0];

    v8f acc0, acc1, acc2, acc3;
    {
      // Peeled K-chunk 0: C operand = bias fragments.
      const int kb = halfsel * 8;
      v8bf lo = *(const v8bf*)(arow_e + kb);
      v8bf hi = *(const v8bf*)(arow_e + kb + 16);
      v16bf a;
      #pragma unroll
      for (int i = 0; i < 8; ++i) { a[i] = lo[i]; a[i + 8] = hi[i]; }
      acc0 = __builtin_amdgcn_wmma_f32_16x16x32_bf16(false, a, false, Bf[0][0], (short)0, biasf0, false, false);
      acc1 = __builtin_amdgcn_wmma_f32_16x16x32_bf16(false, a, false, Bf[1][0], (short)0, biasf1, false, false);
      acc2 = __builtin_amdgcn_wmma_f32_16x16x32_bf16(false, a, false, Bf[2][0], (short)0, biasf2, false, false);
      acc3 = __builtin_amdgcn_wmma_f32_16x16x32_bf16(false, a, false, Bf[3][0], (short)0, biasf3, false, false);
    }
    #pragma unroll
    for (int c = 1; c < 8; ++c) {
      const __bf16* arow = (c < 4) ? arow_e : arow_h;
      const int kb = (c & 3) * 32 + halfsel * 8;
      v8bf lo = *(const v8bf*)(arow + kb);
      v8bf hi = *(const v8bf*)(arow + kb + 16);
      v16bf a;
      #pragma unroll
      for (int i = 0; i < 8; ++i) { a[i] = lo[i]; a[i + 8] = hi[i]; }
      acc0 = __builtin_amdgcn_wmma_f32_16x16x32_bf16(false, a, false, Bf[0][c], (short)0, acc0, false, false);
      acc1 = __builtin_amdgcn_wmma_f32_16x16x32_bf16(false, a, false, Bf[1][c], (short)0, acc1, false, false);
      acc2 = __builtin_amdgcn_wmma_f32_16x16x32_bf16(false, a, false, Bf[2][c], (short)0, acc2, false, false);
      acc3 = __builtin_amdgcn_wmma_f32_16x16x32_bf16(false, a, false, Bf[3][c], (short)0, acc3, false, false);
    }
    __syncthreads();  // BARRIER B: all waves finished reading EmbBuf[t&1]/Hbuf

    // ---- LSTM cell update, register-local per wave ----
    // Fragment element e <-> row = e + 8*halfsel, col = 16w + ln.
    #pragma unroll
    for (int e = 0; e < 8; ++e) {
      float iv = sigm(acc0[e]);
      float fv = sigm(acc1[e]);
      float gv = fast_tanh(acc2[e]);
      float ov = sigm(acc3[e]);
      float cn = fv * cst[e] + iv * gv;
      cst[e] = cn;
      float hv = ov * fast_tanh(cn);
      hf[e] = hv;
      Hbuf[e + halfsel * 8][w * 16 + ln] = f2bf(hv);  // h_{t+1} for next step
    }

    // My async prefetch into EmbBuf[(t+1)&1] must be in LDS before BARRIER A
    // of the next iteration makes it visible.
    if constexpr (USE_EBF) wait_asynccnt0();
  }

  // ---- final linear: out = h_last @ W_l^T + b_l ----
  #pragma unroll
  for (int e = 0; e < 8; ++e) {
    Hlds[e + halfsel * 8][w * 16 + ln] = hf[e];
  }
  __syncthreads();
  if (tid < ROWS * NCLS) {
    int r = tid >> 3, cls = tid & 7;
    float s = b_l[cls];
    #pragma unroll 8
    for (int k = 0; k < HID; ++k) s += Hlds[r][k] * W_l[cls * HID + k];
    out[(g * ROWS + r) * NCLS + cls] = s;
  }
}

extern "C" void kernel_launch(void* const* d_in, const int* in_sizes, int n_in,
                              void* d_out, int out_size, void* d_ws, size_t ws_size,
                              hipStream_t stream) {
  const int*   x    = (const int*)  d_in[0];
  const float* E    = (const float*)d_in[1];
  const float* W_ih = (const float*)d_in[2];
  const float* W_hh = (const float*)d_in[3];
  const float* b_ih = (const float*)d_in[4];
  const float* b_hh = (const float*)d_in[5];
  const float* W_l  = (const float*)d_in[6];
  const float* b_l  = (const float*)d_in[7];
  float*  out = (float*)d_out;
  __bf16* Wc  = (__bf16*)d_ws;
  __bf16* Ebf = (__bf16*)((char*)d_ws + WC_BYTES);

  prep_weights<<<GATES, KTOT, 0, stream>>>(W_ih, W_hh, Wc);

  if (ws_size >= WC_BYTES + EBF_BYTES) {
    int nblk = (int)(((size_t)VOCAB * EMB + 255) / 256);
    prep_embed<<<nblk, 256, 0, stream>>>(E, Ebf);
    lstm_kernel<true><<<NWG, 256, 0, stream>>>(x, E, Ebf, Wc, b_ih, b_hh, W_l, b_l, out);
  } else {
    lstm_kernel<false><<<NWG, 256, 0, stream>>>(x, E, Ebf, Wc, b_ih, b_hh, W_l, b_l, out);
  }
}
